// ParallelDroplessMLP_2302102471516
// MI455X (gfx1250) — compile-verified, matrix operand order checked by reading, not run on hardware
//
#include <hip/hip_runtime.h>
#include <hip/hip_bf16.h>

// Problem constants (match reference)
#define SLEN 2048
#define BSZ  4
#define HSZ  1024
#define FFN  4096
#define NEXP 8
#define TOPK 2
#define NTOK (SLEN*BSZ)        // 8192 tokens
#define NROWS (NTOK*TOPK)      // 16384 routed rows
#define MT 64                  // rows per M-tile (multiple of 16)
#define MAX_MTILES 264         // ceil((NROWS + NEXP*(MT-1))/MT)
#define MAXPAD (MAX_MTILES*MT) // 16896 padded row slots
#define BN 256                 // cols per block tile

typedef __attribute__((ext_vector_type(16))) _Float16 v16h;
typedef __attribute__((ext_vector_type(2)))  _Float16 v2h;
typedef __attribute__((ext_vector_type(8)))  float    v8f;

// Workspace layout (bytes)
#define META_OFF 0
#define PADOFF_OFF 64
#define ROWMAP_OFF 128
#define H_OFF  (ROWMAP_OFF + MAXPAD*4)                 // 67712, 16B aligned
#define Y_OFF  (H_OFF + (size_t)MAXPAD*FFN*2)          // +138412032

// Branch-free tanh-approx gelu: tanh(|u|) = (1-e^{-2|u|})/(1+e^{-2|u|})
__device__ __forceinline__ float gelu_tanh(float v) {
    float u  = 0.7978845608028654f * (v + 0.044715f * v * v * v);
    float au = fabsf(u);
    float ex = __expf(-2.0f * au);
    float th = __builtin_copysignf((1.0f - ex) / (1.0f + ex), u);
    return 0.5f * v * (1.0f + th);
}

// ---------------- routing setup ----------------
__global__ void moe_init(int* meta, int* rowmap) {
    int i = blockIdx.x * 256 + threadIdx.x;
    if (i < 16) meta[i] = 0;
    if (i < MAXPAD) rowmap[i] = -1;
}

__global__ void moe_count(const int* __restrict__ eids, int* counts) {
    int i = blockIdx.x * 256 + threadIdx.x;
    if (i < NROWS) atomicAdd(&counts[eids[i]], 1);
}

__global__ void moe_scan(const int* __restrict__ counts, int* padOff) {
    if (threadIdx.x == 0) {
        int acc = 0;
        padOff[0] = 0;
        for (int e = 0; e < NEXP; ++e) {
            int padded = (counts[e] + MT - 1) & ~(MT - 1);
            acc += padded;
            padOff[e + 1] = acc;
        }
    }
}

__global__ void moe_scatter(const int* __restrict__ eids, int* cursors,
                            const int* __restrict__ padOff, int* rowmap) {
    int i = blockIdx.x * 256 + threadIdx.x;
    if (i < NROWS) {
        int e = eids[i];
        int pos = padOff[e] + atomicAdd(&cursors[e], 1);
        rowmap[pos] = i;
    }
}

// Pack two f32 into LDS as {f16 lo, f16 hi} with one 32-bit store.
__device__ __forceinline__ void pack_store(_Float16* dst, float lo, float hi) {
    v2h p;
    p.x = (_Float16)lo;
    p.y = (_Float16)hi;
    *reinterpret_cast<v2h*>(dst) = p;
}

// ---------------- grouped GEMM1: h = gelu(x_rep @ W1[e]) ----------------
// grid: (FFN/BN, MAX_MTILES), block 256 (8 waves: 2 M-waves x 4 N-waves)
// per wave: 32 rows x 64 cols -> 2 A-frags x 4 B-frags -> 8 WMMA per K-step.
// Software pipelined: next K-slab prefetched into registers while WMMAs run.
__global__ __launch_bounds__(256) void moe_gemm1(
    const float* __restrict__ x, const float* __restrict__ w1,
    const int* __restrict__ padOff, const int* __restrict__ rowmap,
    _Float16* __restrict__ h)
{
    __shared__ _Float16 As[MT][40];      // rows x k (f16, padded)
    __shared__ _Float16 Bs[BN][40];      // n x k (f16, transposed, padded)

    int padTotal = padOff[NEXP];
    int mbase = blockIdx.y * MT;
    if (mbase >= padTotal) return;

    int e = 0;
    while (e < NEXP - 1 && mbase >= padOff[e + 1]) ++e;
    const float* __restrict__ W = w1 + (size_t)e * HSZ * FFN;

    int n0   = blockIdx.x * BN;
    int tid  = threadIdx.x;
    int lane = tid & 31;
    int wv   = tid >> 5;
    int wm   = wv >> 2;      // 0..1  (32 rows each)
    int wn   = wv & 3;       // 0..3  (64 cols each)

    // A staging: row tid>>2, 8 contiguous floats at col (tid&3)*8 (2 float4)
    int arow  = tid >> 2;
    int acol0 = (tid & 3) * 8;
    int rm    = rowmap[mbase + arow];
    const float* __restrict__ asrc =
        (rm >= 0) ? (x + (size_t)(rm >> 1) * HSZ + acol0) : nullptr;
    // B staging: k-pair rows (2*bkp, 2*bkp+1), 16 contiguous n at (tid&15)*16
    int bkp  = tid >> 4;            // 0..15
    int bnc0 = (tid & 15) * 16;     // 0..240
    const float* __restrict__ bsrc = W + (size_t)(2 * bkp) * FFN + n0 + bnc0;

    float4 pa[2], pb0[4], pb1[4];
    #pragma unroll
    for (int i = 0; i < 2; ++i)
        pa[i] = asrc ? *reinterpret_cast<const float4*>(asrc + i * 4)
                     : make_float4(0.f, 0.f, 0.f, 0.f);
    #pragma unroll
    for (int i = 0; i < 4; ++i) {
        pb0[i] = *reinterpret_cast<const float4*>(bsrc + i * 4);
        pb1[i] = *reinterpret_cast<const float4*>(bsrc + FFN + i * 4);
    }

    v8f acc[2][4] = {};

    for (int k0 = 0; k0 < HSZ; k0 += 32) {
        // commit prefetched slab to LDS (f32 -> f16)
        #pragma unroll
        for (int i = 0; i < 2; ++i) {
            pack_store(&As[arow][acol0 + i * 4 + 0], pa[i].x, pa[i].y);
            pack_store(&As[arow][acol0 + i * 4 + 2], pa[i].z, pa[i].w);
        }
        #pragma unroll
        for (int i = 0; i < 4; ++i) {
            pack_store(&Bs[bnc0 + i * 4 + 0][2 * bkp], pb0[i].x, pb1[i].x);
            pack_store(&Bs[bnc0 + i * 4 + 1][2 * bkp], pb0[i].y, pb1[i].y);
            pack_store(&Bs[bnc0 + i * 4 + 2][2 * bkp], pb0[i].z, pb1[i].z);
            pack_store(&Bs[bnc0 + i * 4 + 3][2 * bkp], pb0[i].w, pb1[i].w);
        }
        __syncthreads();

        // prefetch next slab (in flight during WMMAs); hint slab after that
        int kn = k0 + 32;
        if (kn < HSZ) {
            #pragma unroll
            for (int i = 0; i < 2; ++i)
                pa[i] = asrc ? *reinterpret_cast<const float4*>(asrc + kn + i * 4)
                             : make_float4(0.f, 0.f, 0.f, 0.f);
            #pragma unroll
            for (int i = 0; i < 4; ++i) {
                pb0[i] = *reinterpret_cast<const float4*>(bsrc + (size_t)kn * FFN + i * 4);
                pb1[i] = *reinterpret_cast<const float4*>(bsrc + (size_t)(kn + 1) * FFN + i * 4);
            }
            if (kn + 32 < HSZ)
                __builtin_prefetch(bsrc + (size_t)(kn + 32) * FFN, 0, 0);
        }

        // CDNA5 16-bit A layout: lane = M (0..15), half-lane selects K base 0/8
        int ml = lane & 15;
        int kb = (lane >> 4) * 8;
        v16h a[2];
        #pragma unroll
        for (int mi = 0; mi < 2; ++mi) {
            #pragma unroll
            for (int j = 0; j < 8; ++j) {
                a[mi][j]     = As[wm * 32 + mi * 16 + ml][kb + j];
                a[mi][8 + j] = As[wm * 32 + mi * 16 + ml][16 + kb + j];
            }
        }
        #pragma unroll
        for (int t = 0; t < 4; ++t) {
            v16h b;
            #pragma unroll
            for (int j = 0; j < 8; ++j) {
                b[j]     = Bs[wn * 64 + t * 16 + ml][kb + j];
                b[8 + j] = Bs[wn * 64 + t * 16 + ml][16 + kb + j];
            }
            acc[0][t] = __builtin_amdgcn_wmma_f32_16x16x32_f16(
                false, a[0], false, b, (short)0, acc[0][t], false, false);
            acc[1][t] = __builtin_amdgcn_wmma_f32_16x16x32_f16(
                false, a[1], false, b, (short)0, acc[1][t], false, false);
        }
        __syncthreads();
    }

    // epilogue: gelu -> f16 h (C layout: VGPR r, lanes0-15 -> M=r, lanes16-31 -> M=r+8)
    int ml   = lane & 15;
    int radd = (lane >> 4) * 8;
    #pragma unroll
    for (int mi = 0; mi < 2; ++mi) {
        #pragma unroll
        for (int t = 0; t < 4; ++t) {
            int n = n0 + wn * 64 + t * 16 + ml;
            #pragma unroll
            for (int r = 0; r < 8; ++r) {
                int prow = mbase + wm * 32 + mi * 16 + radd + r;
                h[(size_t)prow * FFN + n] = (_Float16)gelu_tanh(acc[mi][t][r]);
            }
        }
    }
}

// ---------------- grouped GEMM2: y = h @ W2[e], scattered by rowmap ----------------
// grid: (HSZ/BN, MAX_MTILES), block 256; same pipelined structure
__global__ __launch_bounds__(256) void moe_gemm2(
    const _Float16* __restrict__ h, const float* __restrict__ w2,
    const int* __restrict__ padOff, const int* __restrict__ rowmap,
    float* __restrict__ y)
{
    __shared__ _Float16 As[MT][40];
    __shared__ _Float16 Bs[BN][40];

    int padTotal = padOff[NEXP];
    int mbase = blockIdx.y * MT;
    if (mbase >= padTotal) return;

    int e = 0;
    while (e < NEXP - 1 && mbase >= padOff[e + 1]) ++e;
    const float* __restrict__ W = w2 + (size_t)e * FFN * HSZ;

    int n0   = blockIdx.x * BN;
    int tid  = threadIdx.x;
    int lane = tid & 31;
    int wv   = tid >> 5;
    int wm   = wv >> 2;
    int wn   = wv & 3;

    // A staging: row tid>>2, 8 contiguous halves (16B) at col (tid&3)*8
    int arow  = tid >> 2;
    int acol0 = (tid & 3) * 8;
    const _Float16* __restrict__ asrc = h + (size_t)(mbase + arow) * FFN + acol0;
    // B staging: k-pair rows, 16 contiguous n
    int bkp  = tid >> 4;
    int bnc0 = (tid & 15) * 16;
    const float* __restrict__ bsrc = W + (size_t)(2 * bkp) * HSZ + n0 + bnc0;

    float4 pa, pb0[4], pb1[4];
    pa = *reinterpret_cast<const float4*>(asrc);
    #pragma unroll
    for (int i = 0; i < 4; ++i) {
        pb0[i] = *reinterpret_cast<const float4*>(bsrc + i * 4);
        pb1[i] = *reinterpret_cast<const float4*>(bsrc + HSZ + i * 4);
    }

    v8f acc[2][4] = {};

    for (int k0 = 0; k0 < FFN; k0 += 32) {
        *reinterpret_cast<float4*>(&As[arow][acol0]) = pa;
        #pragma unroll
        for (int i = 0; i < 4; ++i) {
            pack_store(&Bs[bnc0 + i * 4 + 0][2 * bkp], pb0[i].x, pb1[i].x);
            pack_store(&Bs[bnc0 + i * 4 + 1][2 * bkp], pb0[i].y, pb1[i].y);
            pack_store(&Bs[bnc0 + i * 4 + 2][2 * bkp], pb0[i].z, pb1[i].z);
            pack_store(&Bs[bnc0 + i * 4 + 3][2 * bkp], pb0[i].w, pb1[i].w);
        }
        __syncthreads();

        int kn = k0 + 32;
        if (kn < FFN) {
            pa = *reinterpret_cast<const float4*>(asrc + kn);
            #pragma unroll
            for (int i = 0; i < 4; ++i) {
                pb0[i] = *reinterpret_cast<const float4*>(bsrc + (size_t)kn * HSZ + i * 4);
                pb1[i] = *reinterpret_cast<const float4*>(bsrc + (size_t)(kn + 1) * HSZ + i * 4);
            }
            if (kn + 32 < FFN)
                __builtin_prefetch(bsrc + (size_t)(kn + 32) * HSZ, 0, 0);
        }

        int ml = lane & 15;
        int kb = (lane >> 4) * 8;
        v16h a[2];
        #pragma unroll
        for (int mi = 0; mi < 2; ++mi) {
            #pragma unroll
            for (int j = 0; j < 8; ++j) {
                a[mi][j]     = As[wm * 32 + mi * 16 + ml][kb + j];
                a[mi][8 + j] = As[wm * 32 + mi * 16 + ml][16 + kb + j];
            }
        }
        #pragma unroll
        for (int t = 0; t < 4; ++t) {
            v16h b;
            #pragma unroll
            for (int j = 0; j < 8; ++j) {
                b[j]     = Bs[wn * 64 + t * 16 + ml][kb + j];
                b[8 + j] = Bs[wn * 64 + t * 16 + ml][16 + kb + j];
            }
            acc[0][t] = __builtin_amdgcn_wmma_f32_16x16x32_f16(
                false, a[0], false, b, (short)0, acc[0][t], false, false);
            acc[1][t] = __builtin_amdgcn_wmma_f32_16x16x32_f16(
                false, a[1], false, b, (short)0, acc[1][t], false, false);
        }
        __syncthreads();
    }

    int ml   = lane & 15;
    int radd = (lane >> 4) * 8;
    #pragma unroll
    for (int mi = 0; mi < 2; ++mi) {
        #pragma unroll
        for (int t = 0; t < 4; ++t) {
            int n = n0 + wn * 64 + t * 16 + ml;
            #pragma unroll
            for (int r = 0; r < 8; ++r) {
                int prow = mbase + wm * 32 + mi * 16 + radd + r;
                int rmo = rowmap[prow];
                if (rmo >= 0) y[(size_t)rmo * HSZ + n] = acc[mi][t][r];
            }
        }
    }
}

// ---------------- combine: out[t] = w0*y[2t] + w1*y[2t+1] ----------------
__global__ void moe_combine(const float* __restrict__ y,
                            const float* __restrict__ ew,
                            float* __restrict__ out)
{
    int idx = blockIdx.x * 256 + threadIdx.x;
    if (idx < NTOK * HSZ) {
        int t = idx / HSZ;
        int j = idx - t * HSZ;
        float w0 = ew[t * 2 + 0], w1 = ew[t * 2 + 1];
        out[idx] = w0 * y[(size_t)(t * 2 + 0) * HSZ + j] +
                   w1 * y[(size_t)(t * 2 + 1) * HSZ + j];
    }
}

extern "C" void kernel_launch(void* const* d_in, const int* in_sizes, int n_in,
                              void* d_out, int out_size, void* d_ws, size_t ws_size,
                              hipStream_t stream) {
    const float* x   = (const float*)d_in[0];
    const float* ew  = (const float*)d_in[1];
    const float* w1  = (const float*)d_in[2];
    const float* w2  = (const float*)d_in[3];
    const int*   eid = (const int*)d_in[4];
    float* out = (float*)d_out;

    char* ws = (char*)d_ws;
    int* meta    = (int*)(ws + META_OFF);     // counts[8], cursors[8]
    int* padOff  = (int*)(ws + PADOFF_OFF);   // [9]
    int* rowmap  = (int*)(ws + ROWMAP_OFF);   // [MAXPAD]
    _Float16* hbuf = (_Float16*)(ws + H_OFF);
    float*    ybuf = (float*)(ws + Y_OFF);

    moe_init   <<<(MAXPAD + 255) / 256, 256, 0, stream>>>(meta, rowmap);
    moe_count  <<<(NROWS + 255) / 256, 256, 0, stream>>>(eid, meta);
    moe_scan   <<<1, 32, 0, stream>>>(meta, padOff);
    moe_scatter<<<(NROWS + 255) / 256, 256, 0, stream>>>(eid, meta + 8, padOff, rowmap);

    moe_gemm1<<<dim3(FFN / BN, MAX_MTILES), 256, 0, stream>>>(x, w1, padOff, rowmap, hbuf);
    moe_gemm2<<<dim3(HSZ / BN, MAX_MTILES), 256, 0, stream>>>(hbuf, w2, padOff, rowmap, ybuf);
    moe_combine<<<(NTOK * HSZ + 255) / 256, 256, 0, stream>>>(ybuf, ew, out);
}
